// GATGRU_50242527428898
// MI455X (gfx1250) — matrix operand order repaired and checked by reading, not compiled, not run on hardware
//
#include <hip/hip_runtime.h>
#include <hip/hip_bf16.h>
#include <math.h>

// ---------------------------------------------------------------------------
// Problem constants (fixed by the reference)
// ---------------------------------------------------------------------------
constexpr int N_    = 10000;
constexpr int E_    = 160000;
constexpr int B_    = 4;
constexpr int P_    = 12;
constexpr int HEADS_= 4;
constexpr int DIM_  = 32;
constexpr int GD_   = 128;   // HEADS*DIM
constexpr int GH_   = 64;
constexpr int HOR_  = 6;
constexpr int BN_   = B_ * N_;   // 40000 rows (divisible by 16)

typedef __attribute__((ext_vector_type(16))) _Float16 v16h;
typedef __attribute__((ext_vector_type(8)))  float    v8f;

// ---------------------------------------------------------------------------
// Helpers
// ---------------------------------------------------------------------------
__device__ __forceinline__ unsigned fenc(float f) {
    unsigned u = __float_as_uint(f);
    return (u & 0x80000000u) ? ~u : (u | 0x80000000u);   // order-preserving
}
__device__ __forceinline__ float fdec(unsigned e) {
    return __uint_as_float((e & 0x80000000u) ? (e ^ 0x80000000u) : ~e);
}
__device__ __forceinline__ float leaky02(float x) { return x > 0.0f ? x : 0.2f * x; }
__device__ __forceinline__ float eluf(float x)    { return x > 0.0f ? x : (expf(x) - 1.0f); }
__device__ __forceinline__ float sigm(float x)    { return 1.0f / (1.0f + expf(-x)); }

__device__ __forceinline__ v16h cvt16(float4 x0, float4 x1, float4 x2, float4 x3) {
    v16h v;
    v[0]=(_Float16)x0.x;  v[1]=(_Float16)x0.y;  v[2]=(_Float16)x0.z;  v[3]=(_Float16)x0.w;
    v[4]=(_Float16)x1.x;  v[5]=(_Float16)x1.y;  v[6]=(_Float16)x1.z;  v[7]=(_Float16)x1.w;
    v[8]=(_Float16)x2.x;  v[9]=(_Float16)x2.y;  v[10]=(_Float16)x2.z; v[11]=(_Float16)x2.w;
    v[12]=(_Float16)x3.x; v[13]=(_Float16)x3.y; v[14]=(_Float16)x3.z; v[15]=(_Float16)x3.w;
    return v;
}

// ---------------------------------------------------------------------------
// WMMA GEMM: out[M x N] = A[M x K] * W[N x K]^T (+ bias[N])
// M%16==0, N%(16*NT)==0, K%32==0 (K compile-time -> fully unrolled).
// One wave32 per (16 x 16*NT) output strip. Per K-chunk: A fragment loaded/
// converted once, then ALL NT B fragments are materialized into disjoint
// registers BEFORE the NT back-to-back v_wmma issues -> no WMMA source
// registers are rewritten while a WMMA is in flight, avoiding the 4-NOP
// WMMA->VALU WAR hazard (ISA 7.12.1) per matrix op.
// A layout (16x32 f16): lane<16 -> row M=lane, halves {K..K+7},{K+16..K+23};
//                       lane>=16 -> same rows, halves {K+8..},{K+24..}.
// B layout (32x16 f16): lane -> col N=lane&15; same K halves as A:
//                       lanes<16 {K..K+7},{K+16..K+23}; lanes>=16 {K+8..},{K+24..}.
// C/D: vgpr r, lane l -> row r+8*(l>>4), col l&15.
// ---------------------------------------------------------------------------
template<int K, int NT>
__global__ __launch_bounds__(256)
void wmma_gemm_atw(const float* __restrict__ A, const float* __restrict__ W,
                   const float* __restrict__ bias, float* __restrict__ out,
                   int M, int N, int totalWaves) {
    int wave = (blockIdx.x * blockDim.x + threadIdx.x) >> 5;
    if (wave >= totalWaves) return;
    int lane   = threadIdx.x & 31;
    int groups = (N >> 4) / NT;           // N-strip groups per M-row
    int mt  = wave / groups;
    int nt0 = (wave - mt * groups) * NT;  // first 16-wide N tile of this strip
    int half = lane >> 4;
    int l16  = lane & 15;
    int m = mt * 16 + l16;
    const float* __restrict__ Arow = A + (size_t)m * K;

    const float* Wrow[NT];
    #pragma unroll
    for (int i = 0; i < NT; ++i)
        Wrow[i] = W + (size_t)((nt0 + i) * 16 + l16) * K;

    v8f acc[NT];
    #pragma unroll
    for (int i = 0; i < NT; ++i) acc[i] = (v8f){};

    #pragma unroll
    for (int kc = 0; kc < K; kc += 32) {
        int ka = kc + half * 8;   // A: two 8-float runs at ka and ka+16
        v16h a = cvt16(*(const float4*)(Arow + ka),
                       *(const float4*)(Arow + ka + 4),
                       *(const float4*)(Arow + ka + 16),
                       *(const float4*)(Arow + ka + 20));
        int kb = kc + half * 8;   // B: two 8-float runs at kb and kb+16
        v16h b[NT];
        #pragma unroll
        for (int i = 0; i < NT; ++i) {
            b[i] = cvt16(*(const float4*)(Wrow[i] + kb),
                         *(const float4*)(Wrow[i] + kb + 4),
                         *(const float4*)(Wrow[i] + kb + 16),
                         *(const float4*)(Wrow[i] + kb + 20));
        }
        #pragma unroll
        for (int i = 0; i < NT; ++i) {
            acc[i] = __builtin_amdgcn_wmma_f32_16x16x32_f16(
                         /*neg_a=*/false, a, /*neg_b=*/false, b[i],
                         /*c_mod=*/(short)0, acc[i],
                         /*reuse_a=*/false, /*reuse_b=*/false);
        }
    }

    #pragma unroll
    for (int i = 0; i < NT; ++i) {
        int n  = (nt0 + i) * 16 + l16;
        float bv = bias ? bias[n] : 0.0f;
        #pragma unroll
        for (int r = 0; r < 8; ++r) {
            int row = mt * 16 + r + 8 * half;
            out[(size_t)row * N + n] = acc[i][r] + bv;
        }
    }
}

// ---------------------------------------------------------------------------
// Layer-0 projection: h[b,n,gd] = x[b,t,n,0] * W0[gd]  (C==1)
// ---------------------------------------------------------------------------
__global__ __launch_bounds__(256)
void proj0_kernel(const float* __restrict__ x, const float* __restrict__ W0,
                  float* __restrict__ hfeat, int t) {
    int gid = blockIdx.x * blockDim.x + threadIdx.x;
    if (gid >= BN_ * GD_) return;
    int bn = gid >> 7;          // /GD_
    int gd = gid & (GD_ - 1);
    int b  = bn / N_;
    int n  = bn - b * N_;
    float xv = x[((size_t)(b * P_ + t)) * N_ + n];
    hfeat[gid] = xv * W0[gd];
}

// ---------------------------------------------------------------------------
// Per-node attention pre-dots: sl[bn,h]=<h[bn,h,:],al[h]>, sr likewise.
// ---------------------------------------------------------------------------
__global__ __launch_bounds__(256)
void slsr_kernel(const float* __restrict__ hfeat,
                 const float* __restrict__ al, const float* __restrict__ ar,
                 float* __restrict__ sl, float* __restrict__ sr) {
    int gid = blockIdx.x * blockDim.x + threadIdx.x;
    if (gid >= BN_ * HEADS_) return;
    int bn = gid >> 2;
    int h  = gid & 3;
    const float* hp  = hfeat + (size_t)bn * GD_ + h * DIM_;
    const float* alp = al + h * DIM_;
    const float* arp = ar + h * DIM_;
    float a = 0.0f, r = 0.0f;
    #pragma unroll 8
    for (int d = 0; d < DIM_; ++d) { float v = hp[d]; a += v * alp[d]; r += v * arp[d]; }
    sl[gid] = a; sr[gid] = r;
}

// ---------------------------------------------------------------------------
// Edge pass 1: segment max of leaky(sl[src]+sr[dst]) over dst (encoded u32).
// ---------------------------------------------------------------------------
__global__ __launch_bounds__(256)
void edge_max_kernel(const int* __restrict__ src, const int* __restrict__ dst,
                     const float* __restrict__ sl, const float* __restrict__ sr,
                     unsigned* __restrict__ menc) {
    int gid = blockIdx.x * blockDim.x + threadIdx.x;
    if (gid >= B_ * E_) return;
    int b = gid / E_;
    int e = gid - b * E_;
    int s = src[e], d = dst[e];
    int sb = (b * N_ + s) * HEADS_, db = (b * N_ + d) * HEADS_;
    #pragma unroll
    for (int h = 0; h < HEADS_; ++h) {
        float val = leaky02(sl[sb + h] + sr[db + h]);
        atomicMax(&menc[db + h], fenc(val));
    }
}

// ---------------------------------------------------------------------------
// Edge pass 2: den[dst,h] += exp(e - m[dst,h])
// ---------------------------------------------------------------------------
__global__ __launch_bounds__(256)
void edge_den_kernel(const int* __restrict__ src, const int* __restrict__ dst,
                     const float* __restrict__ sl, const float* __restrict__ sr,
                     const unsigned* __restrict__ menc, float* __restrict__ den) {
    int gid = blockIdx.x * blockDim.x + threadIdx.x;
    if (gid >= B_ * E_) return;
    int b = gid / E_;
    int e = gid - b * E_;
    int s = src[e], d = dst[e];
    int sb = (b * N_ + s) * HEADS_, db = (b * N_ + d) * HEADS_;
    #pragma unroll
    for (int h = 0; h < HEADS_; ++h) {
        float val = leaky02(sl[sb + h] + sr[db + h]);
        float ex  = expf(val - fdec(menc[db + h]));
        atomicAdd(&den[db + h], ex);
    }
}

// ---------------------------------------------------------------------------
// Edge pass 3: agg[dst, h, :] += alpha * hfeat[src, h, :]
// One thread per (b, edge, head); 32 scalar atomics each.
// ---------------------------------------------------------------------------
__global__ __launch_bounds__(256)
void edge_scatter_kernel(const int* __restrict__ src, const int* __restrict__ dst,
                         const float* __restrict__ sl, const float* __restrict__ sr,
                         const unsigned* __restrict__ menc, const float* __restrict__ den,
                         const float* __restrict__ hfeat, float* __restrict__ agg) {
    int gid = blockIdx.x * blockDim.x + threadIdx.x;
    if (gid >= B_ * E_ * HEADS_) return;
    int h  = gid & 3;
    int be = gid >> 2;
    int b  = be / E_;
    int e  = be - b * E_;
    int s = src[e], d = dst[e];
    int sb = (b * N_ + s) * HEADS_ + h, db = (b * N_ + d) * HEADS_ + h;
    float val   = leaky02(sl[sb] + sr[db]);
    float ex    = expf(val - fdec(menc[db]));
    float alpha = ex / (den[db] + 1e-12f);
    const float* hs = hfeat + ((size_t)(b * N_ + s)) * GD_ + h * DIM_;
    float*       op = agg   + ((size_t)(b * N_ + d)) * GD_ + h * DIM_;
    #pragma unroll 8
    for (int dd = 0; dd < DIM_; ++dd) atomicAdd(&op[dd], alpha * hs[dd]);
}

__global__ __launch_bounds__(256)
void elu_kernel(float* __restrict__ p, int count) {
    int gid = blockIdx.x * blockDim.x + threadIdx.x;
    if (gid < count) p[gid] = eluf(p[gid]);
}

// ---------------------------------------------------------------------------
// GRU pointwise update: h = (1-z)*n + z*h
// gi, gh are (BN, 3*GH) row-major [r | z | n blocks of GH].
// ---------------------------------------------------------------------------
__global__ __launch_bounds__(256)
void gru_pointwise_kernel(const float* __restrict__ gi, const float* __restrict__ gh,
                          float* __restrict__ hstate) {
    int gid = blockIdx.x * blockDim.x + threadIdx.x;
    if (gid >= BN_ * GH_) return;
    int row = gid / GH_;
    int j   = gid - row * GH_;
    const float* gir = gi + (size_t)row * 3 * GH_;
    const float* ghr = gh + (size_t)row * 3 * GH_;
    float r  = sigm(gir[j]           + ghr[j]);
    float z  = sigm(gir[GH_ + j]     + ghr[GH_ + j]);
    float nn = tanhf(gir[2 * GH_ + j] + r * ghr[2 * GH_ + j]);
    float hv = hstate[gid];
    hstate[gid] = (1.0f - z) * nn + z * hv;
}

// ---------------------------------------------------------------------------
// Head: out[b, hor, n] = <h[b*N+n], head_w[hor]> + head_b[hor]
// ---------------------------------------------------------------------------
__global__ __launch_bounds__(256)
void head_kernel(const float* __restrict__ hstate, const float* __restrict__ hw,
                 const float* __restrict__ hb, float* __restrict__ out) {
    int gid = blockIdx.x * blockDim.x + threadIdx.x;
    if (gid >= BN_) return;
    int b = gid / N_;
    int n = gid - b * N_;
    const float* hp = hstate + (size_t)gid * GH_;
    #pragma unroll
    for (int hor = 0; hor < HOR_; ++hor) {
        const float* wp = hw + hor * GH_;
        float acc = hb[hor];
        #pragma unroll 8
        for (int j = 0; j < GH_; ++j) acc += hp[j] * wp[j];
        out[((size_t)b * HOR_ + hor) * N_ + n] = acc;
    }
}

// ---------------------------------------------------------------------------
// Launch
// ---------------------------------------------------------------------------
static inline int blks(long long n, int bs = 256) { return (int)((n + bs - 1) / bs); }

extern "C" void kernel_launch(void* const* d_in, const int* in_sizes, int n_in,
                              void* d_out, int out_size, void* d_ws, size_t ws_size,
                              hipStream_t stream) {
    (void)in_sizes; (void)n_in; (void)out_size; (void)ws_size;

    const float* x        = (const float*)d_in[0];
    const int*   edge     = (const int*)  d_in[1];
    const float* W0       = (const float*)d_in[2];
    const float* al0      = (const float*)d_in[3];
    const float* ar0      = (const float*)d_in[4];
    const float* W1       = (const float*)d_in[5];
    const float* al1      = (const float*)d_in[6];
    const float* ar1      = (const float*)d_in[7];
    const float* gru_wih  = (const float*)d_in[8];
    const float* gru_whh  = (const float*)d_in[9];
    const float* gru_bih  = (const float*)d_in[10];
    const float* gru_bhh  = (const float*)d_in[11];
    const float* head_w   = (const float*)d_in[12];
    const float* head_b   = (const float*)d_in[13];
    float* out = (float*)d_out;

    const int* src = edge;
    const int* dst = edge + E_;

    // Workspace layout (floats)
    float* ws = (float*)d_ws;
    float*    hfeat = ws;                                  // BN*GD
    float*    agg   = hfeat + (size_t)BN_ * GD_;           // BN*GD
    float*    gi    = agg   + (size_t)BN_ * GD_;           // BN*3*GH
    float*    gh    = gi    + (size_t)BN_ * 3 * GH_;       // BN*3*GH
    float*    hst   = gh    + (size_t)BN_ * 3 * GH_;       // BN*GH
    float*    sl    = hst   + (size_t)BN_ * GH_;           // BN*HEADS
    float*    sr    = sl    + (size_t)BN_ * HEADS_;        // BN*HEADS
    unsigned* menc  = (unsigned*)(sr + (size_t)BN_ * HEADS_); // BN*HEADS
    float*    den   = (float*)(menc + (size_t)BN_ * HEADS_);  // BN*HEADS

    const size_t featB = (size_t)BN_ * GD_ * sizeof(float);
    const size_t nhB   = (size_t)BN_ * HEADS_ * sizeof(float);

    // h0 = 0 for the GRU
    hipMemsetAsync(hst, 0, (size_t)BN_ * GH_ * sizeof(float), stream);

    const int tilesM = BN_ / 16;   // 2500

    for (int t = 0; t < P_; ++t) {
        // ---- GAT layer 0 ----
        proj0_kernel<<<blks((long long)BN_ * GD_), 256, 0, stream>>>(x, W0, hfeat, t);
        slsr_kernel<<<blks((long long)BN_ * HEADS_), 256, 0, stream>>>(hfeat, al0, ar0, sl, sr);
        hipMemsetAsync(menc, 0, nhB, stream);
        hipMemsetAsync(den,  0, nhB, stream);
        hipMemsetAsync(agg,  0, featB, stream);
        edge_max_kernel<<<blks((long long)B_ * E_), 256, 0, stream>>>(src, dst, sl, sr, menc);
        edge_den_kernel<<<blks((long long)B_ * E_), 256, 0, stream>>>(src, dst, sl, sr, menc, den);
        edge_scatter_kernel<<<blks((long long)B_ * E_ * HEADS_), 256, 0, stream>>>(
            src, dst, sl, sr, menc, den, hfeat, agg);
        elu_kernel<<<blks((long long)BN_ * GD_), 256, 0, stream>>>(agg, BN_ * GD_);

        // ---- GAT layer 1: hfeat = agg @ W1^T (WMMA, K=128, N=128, NT=4) ----
        {
            int tw = tilesM * ((GD_ / 16) / 4);
            wmma_gemm_atw<GD_, 4><<<blks((long long)tw * 32), 256, 0, stream>>>(
                agg, W1, nullptr, hfeat, BN_, GD_, tw);
        }
        slsr_kernel<<<blks((long long)BN_ * HEADS_), 256, 0, stream>>>(hfeat, al1, ar1, sl, sr);
        hipMemsetAsync(menc, 0, nhB, stream);
        hipMemsetAsync(den,  0, nhB, stream);
        hipMemsetAsync(agg,  0, featB, stream);
        edge_max_kernel<<<blks((long long)B_ * E_), 256, 0, stream>>>(src, dst, sl, sr, menc);
        edge_den_kernel<<<blks((long long)B_ * E_), 256, 0, stream>>>(src, dst, sl, sr, menc, den);
        edge_scatter_kernel<<<blks((long long)B_ * E_ * HEADS_), 256, 0, stream>>>(
            src, dst, sl, sr, menc, den, hfeat, agg);
        elu_kernel<<<blks((long long)BN_ * GD_), 256, 0, stream>>>(agg, BN_ * GD_);
        // agg now holds Z_t (BN x GD)

        // ---- GRU step (WMMA GEMMs + pointwise) ----
        {
            int twI = tilesM * (((3 * GH_) / 16) / 4);   // N=192 -> 3 strips of 4 tiles
            wmma_gemm_atw<GD_, 4><<<blks((long long)twI * 32), 256, 0, stream>>>(
                agg, gru_wih, gru_bih, gi, BN_, 3 * GH_, twI);
            wmma_gemm_atw<GH_, 4><<<blks((long long)twI * 32), 256, 0, stream>>>(
                hst, gru_whh, gru_bhh, gh, BN_, 3 * GH_, twI);
        }
        gru_pointwise_kernel<<<blks((long long)BN_ * GH_), 256, 0, stream>>>(gi, gh, hst);
    }

    // ---- Head ----
    head_kernel<<<blks((long long)BN_), 256, 0, stream>>>(hst, head_w, head_b, out);
}
